// BidirectionalCross_10754598109690
// MI455X (gfx1250) — matrix operand order repaired and checked by reading, not compiled, not run on hardware
//
#include <hip/hip_runtime.h>
#include <hip/hip_bf16.h>
#include <math.h>

typedef __bf16 bf16_t;
typedef __attribute__((ext_vector_type(16))) __bf16 bf16x16;
typedef __attribute__((ext_vector_type(8)))  float  f32x8;

#define NB   2
#define LL   3000
#define SSQ  3000
#define CC   256
#define HH   256
#define FCH  8
#define DH   32
#define NF   (NB*FCH)
#define TPAD 3008
#define INV_SCALE 0.17677669529663687f
#define NEG_BIG  (-3.0e38f)

union FragB { bf16x16 v; uint4 q[2]; bf16_t h[16]; };
union AccU  { f32x8  v; float f[8]; };

// ---------------- prep kernels ----------------
__global__ void zero_bf16_kernel(bf16_t* p, size_t count) {
    size_t i = (size_t)blockIdx.x * blockDim.x + threadIdx.x;
    size_t stride = (size_t)gridDim.x * blockDim.x;
    for (; i < count; i += stride) p[i] = (bf16_t)0.0f;
}

// W [R][Ccol] (f32, row-major) -> WT [Ccol][R] (bf16)
__global__ void transpose_to_bf16_kernel(const float* __restrict__ W,
                                         bf16_t* __restrict__ WT, int R, int Ccol) {
    int i = blockIdx.x * blockDim.x + threadIdx.x;
    int total = R * Ccol;
    if (i < total) {
        int r = i / Ccol, c = i % Ccol;
        WT[(size_t)c * R + r] = (bf16_t)W[(size_t)r * Ccol + c];
    }
}

// ---------------- projection: P = X @ Wproj + b, split heads ----------------
// X: [NB][T][CC] f32 ; WT: [2H][CC] bf16 ; Qout: [NF][TPAD][32] bf16 ; VTout: [NF][32][TPAD] bf16
__global__ void proj_kernel(const float* __restrict__ X,
                            const bf16_t* __restrict__ WT,
                            const float* __restrict__ bias,
                            bf16_t* __restrict__ Qout,
                            bf16_t* __restrict__ VTout,
                            int T) {
    int lane = threadIdx.x & 31;
    int wave = threadIdx.x >> 5;
    int hl   = lane >> 4;     // half-wave
    int l16  = lane & 15;

    int tile = blockIdx.x * 4 + wave;          // NB * 188 * 32 tiles total
    int ct = tile & 31;
    int tt = (tile >> 5) % (TPAD / 16);
    int n  = (tile >> 5) / (TPAD / 16);
    int t0 = tt * 16, c0 = ct * 16;

    AccU acc;
    #pragma unroll
    for (int r = 0; r < 8; ++r) acc.f[r] = 0.0f;

    const float* xrow = X + ((size_t)n * T + min(t0 + l16, T - 1)) * CC;
    for (int k0 = 0; k0 < CC; k0 += 32) {
        FragB a;
        #pragma unroll
        for (int j = 0; j < 8; ++j) a.h[j]     = (bf16_t)xrow[k0 + hl * 8 + j];
        #pragma unroll
        for (int j = 0; j < 8; ++j) a.h[8 + j] = (bf16_t)xrow[k0 + 16 + hl * 8 + j];
        FragB b;
        const bf16_t* wrow = WT + (size_t)(c0 + l16) * CC + k0 + hl * 8;
        b.q[0] = *(const uint4*)(wrow);
        b.q[1] = *(const uint4*)(wrow + 16);
        acc.v = __builtin_amdgcn_wmma_f32_16x16x32_bf16(false, a.v, false, b.v,
                                                        (short)0, acc.v, false, false);
    }

    int c = c0 + l16;
    float bs = bias[c];
    int f = (c >> 5) & 7;
    int d = c & 31;
    int nf = n * FCH + f;
    bool isv = (c >= HH);
    #pragma unroll
    for (int r = 0; r < 8; ++r) {
        int t = t0 + r + 8 * hl;
        if (t < T) {
            bf16_t val = (bf16_t)(acc.f[r] + bs);
            if (!isv) Qout[((size_t)nf * TPAD + t) * DH + d] = val;
            else      VTout[((size_t)nf * DH + d) * TPAD + t] = val;
        }
    }
}

// ---------------- flash cross-attention (one direction) ----------------
// Q,K: [NF][TPAD][32] bf16 ; VT: [NF][32][TPAD] bf16 ; mask: [NB][LL][SSQ] u8
// Out: [NF][TPAD][32] f32
__global__ void flash_kernel(const bf16_t* __restrict__ Q,
                             const bf16_t* __restrict__ K,
                             const bf16_t* __restrict__ VT,
                             const unsigned char* __restrict__ mask,
                             int Tq, int Tk, int mrs, int mcs,
                             float* __restrict__ Out) {
    __shared__ bf16_t ldsP[4][16 * 32];
    int lane = threadIdx.x & 31;
    int wave = threadIdx.x >> 5;
    int hl   = lane >> 4;
    int l16  = lane & 15;
    int nf   = blockIdx.y;
    int n    = nf >> 3;
    int row0 = blockIdx.x * 64 + wave * 16;
    const unsigned char* mbase = mask + (size_t)n * LL * SSQ;

    FragB aq;
    {
        const bf16_t* qrow = Q + ((size_t)nf * TPAD + row0 + l16) * DH;
        aq.q[0] = *(const uint4*)(qrow + hl * 16);
        aq.q[1] = *(const uint4*)(qrow + hl * 16 + 8);
    }

    float rmax[8], rsum[8];
    AccU acc0, acc1;
    #pragma unroll
    for (int r = 0; r < 8; ++r) { rmax[r] = NEG_BIG; rsum[r] = 0.f; acc0.f[r] = 0.f; acc1.f[r] = 0.f; }

    for (int s0 = 0; s0 < TPAD; s0 += 32) {
        // score B-fragments: B[k][n] = K[s0+n][k], contiguous per lane
        FragB bk0, bk1;
        {
            const bf16_t* kr0 = K + ((size_t)nf * TPAD + s0 + l16) * DH + hl * 16;
            const bf16_t* kr1 = K + ((size_t)nf * TPAD + s0 + 16 + l16) * DH + hl * 16;
            bk0.q[0] = *(const uint4*)kr0; bk0.q[1] = *(const uint4*)(kr0 + 8);
            bk1.q[0] = *(const uint4*)kr1; bk1.q[1] = *(const uint4*)(kr1 + 8);
        }
        f32x8 z = {0.f, 0.f, 0.f, 0.f, 0.f, 0.f, 0.f, 0.f};
        AccU c0u, c1u;
        c0u.v = __builtin_amdgcn_wmma_f32_16x16x32_bf16(false, aq.v, false, bk0.v, (short)0, z, false, false);
        c1u.v = __builtin_amdgcn_wmma_f32_16x16x32_bf16(false, aq.v, false, bk1.v, (short)0, z, false, false);

        // mask + scale (sim/SCALE with -inf for masked, like the reference)
        float sc0[8], sc1[8];
        int col0 = s0 + l16, col1 = s0 + 16 + l16;
        int cc0 = min(col0, Tk - 1), cc1 = min(col1, Tk - 1);
        bool cv0 = col0 < Tk, cv1 = col1 < Tk;
        #pragma unroll
        for (int r = 0; r < 8; ++r) {
            int row = row0 + r + 8 * hl;
            int rc  = min(row, Tq - 1);
            bool rv = row < Tq;
            bool m0v = rv && cv0 && (mbase[(size_t)rc * mrs + (size_t)cc0 * mcs] != 0);
            bool m1v = rv && cv1 && (mbase[(size_t)rc * mrs + (size_t)cc1 * mcs] != 0);
            sc0[r] = m0v ? c0u.f[r] * INV_SCALE : NEG_BIG;
            sc1[r] = m1v ? c1u.f[r] * INV_SCALE : NEG_BIG;
        }

        // online softmax: row reductions via 16-lane butterflies (rows live per half-wave)
        float p0[8], p1[8];
        #pragma unroll
        for (int r = 0; r < 8; ++r) {
            float rm = fmaxf(sc0[r], sc1[r]);
            rm = fmaxf(rm, __shfl_xor(rm, 1, 32));
            rm = fmaxf(rm, __shfl_xor(rm, 2, 32));
            rm = fmaxf(rm, __shfl_xor(rm, 4, 32));
            rm = fmaxf(rm, __shfl_xor(rm, 8, 32));
            float nmax  = fmaxf(rmax[r], rm);
            float alpha = __expf(rmax[r] - nmax);   // both NEG_BIG -> exp(0)=1 (sums are 0 anyway)
            rmax[r] = nmax;
            float e0 = (sc0[r] <= -1.5e38f) ? 0.f : __expf(sc0[r] - nmax);
            float e1 = (sc1[r] <= -1.5e38f) ? 0.f : __expf(sc1[r] - nmax);
            p0[r] = e0; p1[r] = e1;
            float ps = e0 + e1;
            ps += __shfl_xor(ps, 1, 32);
            ps += __shfl_xor(ps, 2, 32);
            ps += __shfl_xor(ps, 4, 32);
            ps += __shfl_xor(ps, 8, 32);
            rsum[r] = rsum[r] * alpha + ps;
            acc0.f[r] *= alpha; acc1.f[r] *= alpha;
        }

        // re-fragment P (C-layout -> A-layout) through per-wave LDS
        bf16_t* lp = ldsP[wave];
        #pragma unroll
        for (int r = 0; r < 8; ++r) {
            int rt = r + 8 * hl;
            lp[rt * 32 + l16]      = (bf16_t)p0[r];
            lp[rt * 32 + 16 + l16] = (bf16_t)p1[r];
        }
        asm volatile("s_wait_dscnt 0x0" ::: "memory");
        FragB pa;
        pa.q[0] = *(const uint4*)(lp + l16 * 32 + hl * 16);
        pa.q[1] = *(const uint4*)(lp + l16 * 32 + hl * 16 + 8);

        // V B-fragments from transposed V (contiguous along k)
        FragB bv0, bv1;
        const bf16_t* v0r = VT + ((size_t)nf * DH + l16) * TPAD + s0 + hl * 16;
        const bf16_t* v1r = VT + ((size_t)nf * DH + 16 + l16) * TPAD + s0 + hl * 16;
        bv0.q[0] = *(const uint4*)v0r; bv0.q[1] = *(const uint4*)(v0r + 8);
        bv1.q[0] = *(const uint4*)v1r; bv1.q[1] = *(const uint4*)(v1r + 8);

        acc0.v = __builtin_amdgcn_wmma_f32_16x16x32_bf16(false, pa.v, false, bv0.v, (short)0, acc0.v, false, false);
        acc1.v = __builtin_amdgcn_wmma_f32_16x16x32_bf16(false, pa.v, false, bv1.v, (short)0, acc1.v, false, false);
    }

    // normalize (rsum==0 -> 0, matches nan_to_num) and store
    #pragma unroll
    for (int r = 0; r < 8; ++r) {
        int row = row0 + r + 8 * hl;
        float inv = (rsum[r] > 0.f) ? 1.0f / rsum[r] : 0.f;
        float* orow = Out + ((size_t)nf * TPAD + row) * DH;
        orow[l16]      = acc0.f[r] * inv;
        orow[16 + l16] = acc1.f[r] * inv;
    }
}

// ---------------- merge: Out = concat_heads(Mh) @ Wmerge + b ----------------
// Mh: [NB][FCH][TPAD][32] f32 ; WT: [CC][HH] bf16 ; Out: [NB][T][CC] f32
__global__ void merge_kernel(const float* __restrict__ Mh,
                             const bf16_t* __restrict__ WT,
                             const float* __restrict__ bias,
                             float* __restrict__ Out, int T) {
    int lane = threadIdx.x & 31;
    int wave = threadIdx.x >> 5;
    int hl   = lane >> 4;
    int l16  = lane & 15;

    int tile = blockIdx.x * 4 + wave;          // NB * 188 * 16 tiles
    int ct = tile & 15;
    int tt = (tile >> 4) % (TPAD / 16);
    int n  = (tile >> 4) / (TPAD / 16);
    int t0 = tt * 16, c0 = ct * 16;

    AccU acc;
    #pragma unroll
    for (int r = 0; r < 8; ++r) acc.f[r] = 0.0f;

    int trow = t0 + l16;
    for (int k0 = 0; k0 < HH; k0 += 32) {
        int f = k0 >> 5;
        const float* arow = Mh + (((size_t)n * FCH + f) * TPAD + trow) * DH;
        FragB a;
        #pragma unroll
        for (int j = 0; j < 8; ++j) a.h[j]     = (bf16_t)arow[hl * 8 + j];
        #pragma unroll
        for (int j = 0; j < 8; ++j) a.h[8 + j] = (bf16_t)arow[16 + hl * 8 + j];
        FragB b;
        const bf16_t* wrow = WT + (size_t)(c0 + l16) * HH + k0 + hl * 8;
        b.q[0] = *(const uint4*)wrow; b.q[1] = *(const uint4*)(wrow + 16);
        acc.v = __builtin_amdgcn_wmma_f32_16x16x32_bf16(false, a.v, false, b.v,
                                                        (short)0, acc.v, false, false);
    }
    int c = c0 + l16;
    float bs = bias[c];
    #pragma unroll
    for (int r = 0; r < 8; ++r) {
        int t = t0 + r + 8 * hl;
        if (t < T) Out[((size_t)n * T + t) * CC + c] = acc.f[r] + bs;
    }
}

// ---------------- launch ----------------
extern "C" void kernel_launch(void* const* d_in, const int* in_sizes, int n_in,
                              void* d_out, int out_size, void* d_ws, size_t ws_size,
                              hipStream_t stream) {
    const float* x0 = (const float*)d_in[0];
    const float* x1 = (const float*)d_in[1];
    const unsigned char* mask = (const unsigned char*)d_in[2];
    const float* Wproj  = (const float*)d_in[3];
    const float* bproj  = (const float*)d_in[4];
    const float* Wmerge = (const float*)d_in[5];
    const float* bmerge = (const float*)d_in[6];
    float* out = (float*)d_out;

    char* ws = (char*)d_ws;
    size_t off = 0;
    auto carve = [&](size_t bytes) -> void* {
        void* p = ws + off;
        off = (off + bytes + 255) & ~(size_t)255;
        return p;
    };
    const size_t qv = (size_t)NF * TPAD * DH;         // elems per q/v buffer
    bf16_t* q0  = (bf16_t*)carve(qv * 2);
    bf16_t* q1  = (bf16_t*)carve(qv * 2);
    bf16_t* v0t = (bf16_t*)carve(qv * 2);
    bf16_t* v1t = (bf16_t*)carve(qv * 2);
    bf16_t* WTp = (bf16_t*)carve((size_t)512 * 256 * 2);
    bf16_t* WTm = (bf16_t*)carve((size_t)256 * 256 * 2);
    float*  m0h = (float*)carve(qv * 4);
    float*  m1h = (float*)carve(qv * 4);

    // zero q/v (covers pad rows; buffers are contiguous: q0,q1,v0t,v1t)
    zero_bf16_kernel<<<1024, 256, 0, stream>>>(q0, qv * 4);
    transpose_to_bf16_kernel<<<(256 * 512 + 255) / 256, 256, 0, stream>>>(Wproj, WTp, 256, 512);
    transpose_to_bf16_kernel<<<(256 * 256 + 255) / 256, 256, 0, stream>>>(Wmerge, WTm, 256, 256);

    // projections (both sides share Wproj)
    proj_kernel<<<(NB * (TPAD / 16) * 32) / 4, 128, 0, stream>>>(x0, WTp, bproj, q0, v0t, LL);
    proj_kernel<<<(NB * (TPAD / 16) * 32) / 4, 128, 0, stream>>>(x1, WTp, bproj, q1, v1t, SSQ);

    // flash attention, both directions (direction 1 = transposed mask, swapped roles)
    dim3 fg(TPAD / 64, NF);
    flash_kernel<<<fg, 128, 0, stream>>>(q0, q1, v1t, mask, LL, SSQ, SSQ, 1, m0h);
    flash_kernel<<<fg, 128, 0, stream>>>(q1, q0, v0t, mask, SSQ, LL, 1, SSQ, m1h);

    // merge + bias
    int mblocks = (NB * (TPAD / 16) * (CC / 16)) / 4;
    merge_kernel<<<mblocks, 128, 0, stream>>>(m0h, WTm, bmerge, out, LL);
    merge_kernel<<<mblocks, 128, 0, stream>>>(m1h, WTm, bmerge, out + (size_t)NB * LL * CC, SSQ);
}
